// CausalSelfAttention_83854941487728
// MI455X (gfx1250) — compile-verified, matrix operand order checked by reading, not run on hardware
//
#include <hip/hip_runtime.h>
#include <hip/hip_bf16.h>

// CDNA5 / gfx1250 causal self-attention.
// bf16 WMMA for all four GEMMs; double-buffered TDM (tensor_load_to_lds)
// stages K/V tiles overlapped with WMMA compute.
// Shapes fixed by reference: b=4, s=2048, d=1024, h=16, hd=64.

typedef __bf16 bf16_t;
typedef __attribute__((ext_vector_type(16))) __bf16 v16bf;
typedef __attribute__((ext_vector_type(8)))  __bf16 v8bf;
typedef __attribute__((ext_vector_type(2)))  __bf16 v2bf;
typedef __attribute__((ext_vector_type(8)))  float  v8f;
typedef unsigned int u32x4 __attribute__((ext_vector_type(4)));
typedef int          i32x4 __attribute__((ext_vector_type(4)));
typedef int          i32x8 __attribute__((ext_vector_type(8)));

#define DIM   1024
#define SEQ   2048
#define BATCH 4
#define HEADS 16
#define HD    64
#define BH    (BATCH*HEADS)      // 64
#define ROWS  (BATCH*SEQ)        // 8192

#define BM 128
#define BN 64
#define BK 32

static __device__ __forceinline__ v16bf combine16(v8bf lo, v8bf hi) {
  v16bf r;
#pragma unroll
  for (int i = 0; i < 8; ++i) { r[i] = lo[i]; r[8 + i] = hi[i]; }
  return r;
}

static __device__ __forceinline__ v8f wmma_bf16(v16bf a, v16bf b, v8f c) {
  // D = A(16x32 bf16) x B(32x16 bf16) + C(16x16 f32)
  return __builtin_amdgcn_wmma_f32_16x16x32_bf16(
      false, a, false, b, (short)0, c, false, false);
}

// ---------------------------------------------------------------------------
// Tensor Data Mover: 2D bf16 tile load, global -> LDS (CDNA5 ISA).
// D# group0: count=1, lds_addr, global_addr[56:0], type=2.
// D# group1: data_size=1 (2B), tensor_dim0/1, tile_dim0/1, dim0 stride.
// Runtime-varying dwords go through readfirstlane so the descriptor is
// provably wave-uniform (no waterfall around the builtin).
// ---------------------------------------------------------------------------
#if __has_builtin(__builtin_amdgcn_tensor_load_to_lds)
#define HAVE_TDM 1
#endif

static __device__ __forceinline__ unsigned rfl(unsigned x) {
  return (unsigned)__builtin_amdgcn_readfirstlane((int)x);
}

static __device__ __forceinline__ void tdm_load_2d_bf16(
    const bf16_t* gptr, unsigned lds_byte_off,
    unsigned tile_d0, unsigned tile_d1,
    unsigned tensor_d0, unsigned tensor_d1, unsigned stride0_elems) {
#ifdef HAVE_TDM
  unsigned long long ga = (unsigned long long)(uintptr_t)gptr;
  u32x4 g0;
  g0[0] = 1u;                                       // count=1 (valid user D#)
  g0[1] = rfl(lds_byte_off);                        // lds_addr
  g0[2] = rfl((unsigned)(ga & 0xFFFFFFFFu));        // global_addr[31:0]
  g0[3] = rfl((unsigned)((ga >> 32) & 0x01FFFFFFu)  // global_addr[56:32]
              | (2u << 30));                        // type=2 ("image")
  i32x8 g1;
  g1[0] = (int)(1u << 16);                          // data_size=1 (2B), no mcast
  g1[1] = (int)((tensor_d0 & 0xFFFFu) << 16);                 // dim0[15:0]
  g1[2] = (int)((tensor_d0 >> 16) | ((tensor_d1 & 0xFFFFu) << 16));
  g1[3] = (int)((tensor_d1 >> 16) | (tile_d0 << 16));         // tile_dim0
  g1[4] = (int)(tile_d1 & 0xFFFFu);                           // tile_dim1
  g1[5] = (int)stride0_elems;                                 // dim0_stride lo
  g1[6] = 0;
  g1[7] = 0;
  i32x4 z4 = {0, 0, 0, 0};
#if __has_include(<hip/amd_detail/amd_gfx1250_TDM.h>)
  i32x8 z8 = {0, 0, 0, 0, 0, 0, 0, 0};
  __builtin_amdgcn_tensor_load_to_lds(g0, g1, z4, z4, z8, 0);   // 6-arg lane
#else
  __builtin_amdgcn_tensor_load_to_lds(g0, g1, z4, z4, 0);       // 5-arg ROCm 7.2
#endif
#else
  (void)gptr; (void)lds_byte_off; (void)tile_d0; (void)tile_d1;
  (void)tensor_d0; (void)tensor_d1; (void)stride0_elems;
#endif
}

// ---------------------------------------------------------------------------
// Kernel 1: QKV GEMM. qkv = x[8192,1024] @ w_qkv[1024,3072]; writes bf16
// Q,K as [b*h][s][64] and V transposed as [b*h][64][s] (K-contiguous for PV).
// ---------------------------------------------------------------------------
__global__ __launch_bounds__(128)
void qkv_gemm_kernel(const float* __restrict__ X, const float* __restrict__ W,
                     bf16_t* __restrict__ Q, bf16_t* __restrict__ K,
                     bf16_t* __restrict__ Vt) {
  __shared__ __align__(16) bf16_t lA[BM][BK];  // [m][k] K-contiguous
  __shared__ __align__(16) bf16_t lB[BN][BK];  // [n][k] K-contiguous

  const int tid  = threadIdx.x;
  const int wave = tid >> 5;
  const int lane = tid & 31;
  const int half = lane >> 4;
  const int l16  = lane & 15;
  const int kb   = half * 8;
  const int m0   = blockIdx.x * BM;
  const int n0   = blockIdx.y * BN;

  v8f acc[2][4] = {};

  for (int k0 = 0; k0 < DIM; k0 += BK) {
#pragma unroll 4
    for (int i = tid; i < (BM * BK) / 2; i += 128) {   // pairwise cvt_pk path
      int m = i >> 4, k = (i & 15) * 2;
      const float2 f = *(const float2*)&X[(size_t)(m0 + m) * DIM + k0 + k];
      v2bf p; p[0] = (bf16_t)f.x; p[1] = (bf16_t)f.y;
      *(v2bf*)&lA[m][k] = p;
    }
#pragma unroll 4
    for (int i = tid; i < BN * BK; i += 128) {         // transpose: scalar
      int n = i >> 5, k = i & 31;
      lB[n][k] = (bf16_t)W[(size_t)(k0 + k) * (3 * DIM) + n0 + n];
    }
    if (k0 + BK < DIM)
      __builtin_prefetch(&X[(size_t)(m0 + tid) * DIM + k0 + BK], 0, 1);
    __syncthreads();

    v16bf af[2];
#pragma unroll
    for (int mt = 0; mt < 2; ++mt)
      af[mt] = combine16(*(const v8bf*)&lA[wave * 32 + mt * 16 + l16][kb],
                         *(const v8bf*)&lA[wave * 32 + mt * 16 + l16][16 + kb]);
#pragma unroll
    for (int t = 0; t < 4; ++t) {
      v16bf bfrag = combine16(*(const v8bf*)&lB[t * 16 + l16][kb],
                              *(const v8bf*)&lB[t * 16 + l16][16 + kb]);
#pragma unroll
      for (int mt = 0; mt < 2; ++mt)
        acc[mt][t] = wmma_bf16(af[mt], bfrag, acc[mt][t]);
    }
    __syncthreads();
  }

#pragma unroll
  for (int mt = 0; mt < 2; ++mt)
#pragma unroll
    for (int t = 0; t < 4; ++t)
#pragma unroll
      for (int r = 0; r < 8; ++r) {
        int grow = m0 + wave * 32 + mt * 16 + r + 8 * half;  // 0..8191
        int gcol = n0 + t * 16 + l16;                        // 0..3071
        int mat  = gcol >> 10;                               // 0=Q 1=K 2=V
        int c    = gcol & (DIM - 1);
        int h = c >> 6, hd = c & 63;
        int b = grow >> 11, s = grow & (SEQ - 1);
        size_t bhh = (size_t)(b * HEADS + h);
        bf16_t val = (bf16_t)acc[mt][t][r];
        if (mat == 0)      Q[(bhh * SEQ + s) * HD + hd] = val;
        else if (mat == 1) K[(bhh * SEQ + s) * HD + hd] = val;
        else               Vt[(bhh * HD + hd) * SEQ + s] = val;   // transposed
      }
}

// ---------------------------------------------------------------------------
// Kernel 2: causal flash attention. 4 waves/block share one (b,h); TDM
// double-buffers 32-key K (32x64) and Vt (64x32) tiles in LDS, prefetching
// chunk ch+1 while WMMAs consume chunk ch (s_wait_tensorcnt 2). Each wave
// owns a 16-query tile with online softmax; P reshaped through per-wave LDS.
// ---------------------------------------------------------------------------
__global__ __launch_bounds__(128)
void attn_kernel(const bf16_t* __restrict__ Q, const bf16_t* __restrict__ K,
                 const bf16_t* __restrict__ Vt, bf16_t* __restrict__ O) {
  const int bh   = blockIdx.x;               // 0..63
  const int tid  = threadIdx.x;
  const int wave = tid >> 5;
  const int lane = tid & 31;
  const int half = lane >> 4;
  const int l16  = lane & 15;
  const int kb   = half * 8;
  const int qt   = blockIdx.y * 4 + wave;    // query tile 0..127

  __shared__ __align__(16) bf16_t ldsK[2][32][HD];     // [buf][key][hd]
  __shared__ __align__(16) bf16_t ldsV[2][HD][32];     // [buf][hd][key]
  __shared__ __align__(16) bf16_t ldsP[4][16][32];     // per-wave P [row][key]

  const bf16_t* qp = Q  + (size_t)bh * SEQ * HD;
  const bf16_t* kp = K  + (size_t)bh * SEQ * HD;
  const bf16_t* vp = Vt + (size_t)bh * HD * SEQ;

  const bf16_t* qrow = qp + (size_t)(qt * 16 + l16) * HD;
  v16bf qf0 = combine16(*(const v8bf*)(qrow + kb),      *(const v8bf*)(qrow + 16 + kb));
  v16bf qf1 = combine16(*(const v8bf*)(qrow + 32 + kb), *(const v8bf*)(qrow + 48 + kb));

  v8f   o[4] = {};
  float m_run[8], l_run[8];
#pragma unroll
  for (int r = 0; r < 8; ++r) { m_run[r] = -1e30f; l_run[r] = 0.0f; }

  const float scale = 0.125f;  // 1/sqrt(64)
  // Block-uniform causal bound (max over the 4 q-tiles in this block).
  const int nchunk = ((blockIdx.y * 4 + 3) * 16 + 16 + 31) >> 5;

#ifdef HAVE_TDM
  if (wave == 0) {  // prologue: stage chunk 0 into buffer 0
    tdm_load_2d_bf16(kp, (unsigned)(uintptr_t)(void*)&ldsK[0][0][0],
                     HD, 32, HD, SEQ, HD);
    tdm_load_2d_bf16(vp, (unsigned)(uintptr_t)(void*)&ldsV[0][0][0],
                     32, HD, SEQ, HD, SEQ);
  }
#endif

  for (int ch = 0; ch < nchunk; ++ch) {
    const int key0 = ch * 32;
    const int cur  = ch & 1;
#ifdef HAVE_TDM
    if (wave == 0) {
      if (ch + 1 < nchunk) {  // prefetch next chunk into the other buffer
        const int nb = (ch + 1) & 1;
        tdm_load_2d_bf16(kp + (size_t)(key0 + 32) * HD,
                         (unsigned)(uintptr_t)(void*)&ldsK[nb][0][0],
                         HD, 32, HD, SEQ, HD);
        tdm_load_2d_bf16(vp + key0 + 32,
                         (unsigned)(uintptr_t)(void*)&ldsV[nb][0][0],
                         32, HD, SEQ, HD, SEQ);
        __builtin_amdgcn_s_wait_tensorcnt(2);  // current pair done; 2 in flight
      } else {
        __builtin_amdgcn_s_wait_tensorcnt(0);
      }
    }
#else
    for (int i = tid; i < 32 * HD; i += 128) {
      ldsK[cur][i >> 6][i & 63] = kp[(size_t)(key0 + (i >> 6)) * HD + (i & 63)];
      ldsV[cur][i >> 5][i & 31] = vp[(size_t)(i >> 5) * SEQ + key0 + (i & 31)];
    }
#endif
    __syncthreads();  // staged tiles visible to all waves

    v8f s0 = {}, s1 = {};
    {  // keys key0..key0+15
      v16bf kf0 = combine16(*(const v8bf*)&ldsK[cur][l16][kb],
                            *(const v8bf*)&ldsK[cur][l16][16 + kb]);
      v16bf kf1 = combine16(*(const v8bf*)&ldsK[cur][l16][32 + kb],
                            *(const v8bf*)&ldsK[cur][l16][48 + kb]);
      s0 = wmma_bf16(qf0, kf0, s0);
      s0 = wmma_bf16(qf1, kf1, s0);
    }
    {  // keys key0+16..key0+31
      v16bf kf0 = combine16(*(const v8bf*)&ldsK[cur][16 + l16][kb],
                            *(const v8bf*)&ldsK[cur][16 + l16][16 + kb]);
      v16bf kf1 = combine16(*(const v8bf*)&ldsK[cur][16 + l16][32 + kb],
                            *(const v8bf*)&ldsK[cur][16 + l16][48 + kb]);
      s1 = wmma_bf16(qf0, kf0, s1);
      s1 = wmma_bf16(qf1, kf1, s1);
    }

    // Mask + scale + online softmax (row = qt*16 + r + 8*half, col = key).
#pragma unroll
    for (int r = 0; r < 8; ++r) {
      const int qg = qt * 16 + r + 8 * half;
      s0[r] = (key0 + l16      <= qg) ? s0[r] * scale : -1e30f;
      s1[r] = (key0 + 16 + l16 <= qg) ? s1[r] * scale : -1e30f;

      float mx = fmaxf(s0[r], s1[r]);
      for (int off = 1; off < 16; off <<= 1)
        mx = fmaxf(mx, __shfl_xor(mx, off, 32));
      const float mnew  = fmaxf(m_run[r], mx);
      const float alpha = __expf(m_run[r] - mnew);
      const float p0 = __expf(s0[r] - mnew);
      const float p1 = __expf(s1[r] - mnew);
      float rs = p0 + p1;
      for (int off = 1; off < 16; off <<= 1)
        rs += __shfl_xor(rs, off, 32);
      l_run[r] = l_run[r] * alpha + rs;
      m_run[r] = mnew;
      s0[r] = p0;
      s1[r] = p1;
#pragma unroll
      for (int t = 0; t < 4; ++t) o[t][r] *= alpha;
    }

    // C-layout -> A-fragment reshape through this wave's LDS tile.
#pragma unroll
    for (int r = 0; r < 8; ++r) {
      const int row = r + 8 * half;
      ldsP[wave][row][l16]      = (bf16_t)s0[r];
      ldsP[wave][row][16 + l16] = (bf16_t)s1[r];
    }
    v16bf pf = combine16(*(const v8bf*)&ldsP[wave][l16][kb],
                         *(const v8bf*)&ldsP[wave][l16][16 + kb]);

    // o += P(16x32) x V(32x64); V fragments K-contiguous from ldsV.
#pragma unroll
    for (int t = 0; t < 4; ++t) {
      const int n = t * 16 + l16;
      v16bf vf = combine16(*(const v8bf*)&ldsV[cur][n][kb],
                           *(const v8bf*)&ldsV[cur][n][16 + kb]);
      o[t] = wmma_bf16(pf, vf, o[t]);
    }
    __syncthreads();  // buffer cur fully consumed before next overwrite
  }

  // Normalize and store attention output, layout [bh][s][64].
#pragma unroll
  for (int r = 0; r < 8; ++r) {
    const float inv = 1.0f / l_run[r];
    const int   row = qt * 16 + r + 8 * half;
#pragma unroll
    for (int t = 0; t < 4; ++t)
      O[((size_t)bh * SEQ + row) * HD + t * 16 + l16] = (bf16_t)(o[t][r] * inv);
  }
}

// ---------------------------------------------------------------------------
// Kernel 3: output projection. out[8192,1024] = attn([b,h,s,hd] -> [b,s,c])
// @ w_proj[1024,1024], fp32 result.
// ---------------------------------------------------------------------------
__global__ __launch_bounds__(128)
void proj_gemm_kernel(const bf16_t* __restrict__ A, const float* __restrict__ W,
                      float* __restrict__ Out) {
  __shared__ __align__(16) bf16_t lA[BM][BK];
  __shared__ __align__(16) bf16_t lB[BN][BK];

  const int tid  = threadIdx.x;
  const int wave = tid >> 5;
  const int lane = tid & 31;
  const int half = lane >> 4;
  const int l16  = lane & 15;
  const int kb   = half * 8;
  const int m0   = blockIdx.x * BM;
  const int n0   = blockIdx.y * BN;

  v8f acc[2][4] = {};

  for (int k0 = 0; k0 < DIM; k0 += BK) {
#pragma unroll 4
    for (int i = tid; i < (BM * BK) / 2; i += 128) {
      int m = i >> 4, k = (i & 15) * 2;
      int grow = m0 + m, c = k0 + k;
      int b = grow >> 11, s = grow & (SEQ - 1);
      int h = c >> 6, hd = c & 63;   // pair never crosses a head (k even)
      *(v2bf*)&lA[m][k] =
          *(const v2bf*)&A[(((size_t)(b * HEADS + h)) * SEQ + s) * HD + hd];
    }
#pragma unroll 4
    for (int i = tid; i < BN * BK; i += 128) {
      int n = i >> 5, k = i & 31;
      lB[n][k] = (bf16_t)W[(size_t)(k0 + k) * DIM + n0 + n];
    }
    __syncthreads();

    v16bf af[2];
#pragma unroll
    for (int mt = 0; mt < 2; ++mt)
      af[mt] = combine16(*(const v8bf*)&lA[wave * 32 + mt * 16 + l16][kb],
                         *(const v8bf*)&lA[wave * 32 + mt * 16 + l16][16 + kb]);
#pragma unroll
    for (int t = 0; t < 4; ++t) {
      v16bf bfrag = combine16(*(const v8bf*)&lB[t * 16 + l16][kb],
                              *(const v8bf*)&lB[t * 16 + l16][16 + kb]);
#pragma unroll
      for (int mt = 0; mt < 2; ++mt)
        acc[mt][t] = wmma_bf16(af[mt], bfrag, acc[mt][t]);
    }
    __syncthreads();
  }

#pragma unroll
  for (int mt = 0; mt < 2; ++mt)
#pragma unroll
    for (int t = 0; t < 4; ++t)
#pragma unroll
      for (int r = 0; r < 8; ++r) {
        int grow = m0 + wave * 32 + mt * 16 + r + 8 * half;
        int gcol = n0 + t * 16 + l16;
        Out[(size_t)grow * DIM + gcol] = acc[mt][t][r];
      }
}

// ---------------------------------------------------------------------------
extern "C" void kernel_launch(void* const* d_in, const int* in_sizes, int n_in,
                              void* d_out, int out_size, void* d_ws, size_t ws_size,
                              hipStream_t stream) {
  (void)in_sizes; (void)n_in; (void)out_size; (void)ws_size;
  const float* x      = (const float*)d_in[0];
  const float* w_qkv  = (const float*)d_in[1];
  const float* w_proj = (const float*)d_in[2];
  float*       out    = (float*)d_out;

  const size_t per = (size_t)BH * SEQ * HD;  // 8,388,608 bf16 elements (16 MB)
  bf16_t* Q  = (bf16_t*)d_ws;
  bf16_t* K  = Q + per;
  bf16_t* Vt = K + per;
  bf16_t* AO = Vt + per;

  qkv_gemm_kernel<<<dim3(ROWS / BM, (3 * DIM) / BN), 128, 0, stream>>>(x, w_qkv, Q, K, Vt);
  attn_kernel<<<dim3(BH, (SEQ / 16) / 4), 128, 0, stream>>>(Q, K, Vt, AO);
  proj_gemm_kernel<<<dim3(ROWS / BM, DIM / BN), 128, 0, stream>>>(AO, w_proj, out);
}
